// _TopKRouter_45878840656611
// MI455X (gfx1250) — matrix operand order repaired and checked by reading, not compiled, not run on hardware
//
#include <hip/hip_runtime.h>

typedef __attribute__((ext_vector_type(16))) __bf16 bf16x16;
typedef __attribute__((ext_vector_type(8)))  float  f32x8;
typedef __attribute__((ext_vector_type(4)))  float  f32x4;

#define DM   4096   // d_model
#define NE   64     // experts
#define NTOK 16384  // B*S
#define KB   (DM / 32)   // 128 K-blocks of 32
#define KSPLIT 4         // K-dimension split across waves
#define MT_PER_BLOCK 2   // M-tiles (16 tokens each) per workgroup

// ---------------------------------------------------------------------------
// Prep: W [E,D] fp32 -> bf16, pre-swizzled into this kernel's contiguous-k WMMA B-fragment
// layout: flat index = ((kb*4 + nt)*32 + lane)*16 + j
//   lanes 0-15 : N = lane,    K = kb*32 + j
//   lanes 16-31: N = lane-16, K = kb*32 + 16 + j
// ---------------------------------------------------------------------------
__global__ __launch_bounds__(256) void wprep_kernel(const float* __restrict__ W,
                                                    __bf16* __restrict__ Wb) {
    int idx  = blockIdx.x * 256 + threadIdx.x;
    int j    = idx & 15;
    int lane = (idx >> 4) & 31;
    int nt   = (idx >> 9) & 3;
    int kb   = idx >> 11;
    int e = nt * 16 + (lane & 15);
    int d = kb * 32 + ((lane >> 4) << 4) + j;
    Wb[idx] = (__bf16)W[e * DM + d];
}

__device__ __forceinline__ void top2_insert(float v, int i,
                                            float& v1, int& i1,
                                            float& v2, int& i2) {
    if (v > v1 || (v == v1 && i < i1)) { v2 = v1; i2 = i1; v1 = v; i1 = i; }
    else if (v > v2 || (v == v2 && i < i2)) { v2 = v; i2 = i; }
}

__device__ __forceinline__ bf16x16 load_bfrag(const __bf16* p) {
    return *(const bf16x16*)p;
}

// ---------------------------------------------------------------------------
// Main: workgroup = 8 waves = 2 M-tiles x 4 K-splits.
// Each wave: 16 tokens x 64 experts over 32 K-blocks (bf16 WMMA, f32 accum).
// Partial logits reduced through LDS; K-split-0 waves run fused softmax+top2.
// ---------------------------------------------------------------------------
__global__ __launch_bounds__(256) void router_kernel(const float* __restrict__ x,
                                                     const __bf16* __restrict__ Wb,
                                                     float* __restrict__ probs,
                                                     float* __restrict__ idx_out,
                                                     float* __restrict__ val_out) {
    __shared__ float red[MT_PER_BLOCK][KSPLIT][4][32 * 8];   // 32 KB

    const int lane = threadIdx.x & 31;
    const int wave = threadIdx.x >> 5;
    const int mt   = wave & (MT_PER_BLOCK - 1);   // M-tile within block
    const int h    = wave >> 1;                   // K-split id 0..3
    const int mrow = lane & 15;
    const int hi   = lane >> 4;
    const int m0   = (blockIdx.x * MT_PER_BLOCK + mt) * 16;

    // A-fragment (this kernel's contiguous k order): lane holds row M = lane%16,
    // K in {koff..koff+15}, koff = (lane<16 ? 0 : 16)
    const float* xrow = x + (size_t)(m0 + mrow) * DM;
    const int koff = hi << 4;
    const int kb0  = h * (KB / KSPLIT);
    const int kb1  = kb0 + (KB / KSPLIT);

    f32x8 acc0 = {}, acc1 = {}, acc2 = {}, acc3 = {};

    // software-pipelined B fragments (one K-step ahead)
    const __bf16* wb = Wb + ((size_t)kb0 << 11) + (lane << 4);
    bf16x16 B0 = load_bfrag(wb);
    bf16x16 B1 = load_bfrag(wb + 512);
    bf16x16 B2 = load_bfrag(wb + 1024);
    bf16x16 B3 = load_bfrag(wb + 1536);

    for (int kb = kb0; kb < kb1; ++kb) {
        const int k0 = kb * 32 + koff;
        // 268MB single-use x stream: non-temporal so it can't evict W from L2
        f32x4 a0 = __builtin_nontemporal_load((const f32x4*)(xrow + k0));
        f32x4 a1 = __builtin_nontemporal_load((const f32x4*)(xrow + k0 + 4));
        f32x4 a2 = __builtin_nontemporal_load((const f32x4*)(xrow + k0 + 8));
        f32x4 a3 = __builtin_nontemporal_load((const f32x4*)(xrow + k0 + 12));

        // prefetch next K-step's B fragments (branchless wrap keeps it in-bounds)
        const __bf16* wbn = Wb + ((size_t)((kb + 1) & (KB - 1)) << 11) + (lane << 4);
        bf16x16 nB0 = load_bfrag(wbn);
        bf16x16 nB1 = load_bfrag(wbn + 512);
        bf16x16 nB2 = load_bfrag(wbn + 1024);
        bf16x16 nB3 = load_bfrag(wbn + 1536);

        bf16x16 A;
        A[0]  = (__bf16)a0.x; A[1]  = (__bf16)a0.y; A[2]  = (__bf16)a0.z; A[3]  = (__bf16)a0.w;
        A[4]  = (__bf16)a1.x; A[5]  = (__bf16)a1.y; A[6]  = (__bf16)a1.z; A[7]  = (__bf16)a1.w;
        A[8]  = (__bf16)a2.x; A[9]  = (__bf16)a2.y; A[10] = (__bf16)a2.z; A[11] = (__bf16)a2.w;
        A[12] = (__bf16)a3.x; A[13] = (__bf16)a3.y; A[14] = (__bf16)a3.z; A[15] = (__bf16)a3.w;

        acc0 = __builtin_amdgcn_wmma_f32_16x16x32_bf16(false, A, false, B0, (short)0, acc0, false, false);
        acc1 = __builtin_amdgcn_wmma_f32_16x16x32_bf16(false, A, false, B1, (short)0, acc1, false, false);
        acc2 = __builtin_amdgcn_wmma_f32_16x16x32_bf16(false, A, false, B2, (short)0, acc2, false, false);
        acc3 = __builtin_amdgcn_wmma_f32_16x16x32_bf16(false, A, false, B3, (short)0, acc3, false, false);

        B0 = nB0; B1 = nB1; B2 = nB2; B3 = nB3;
    }

    // ---- cross-wave K reduction through LDS ----
    if (h != 0) {
        *(f32x8*)&red[mt][h][0][lane * 8] = acc0;
        *(f32x8*)&red[mt][h][1][lane * 8] = acc1;
        *(f32x8*)&red[mt][h][2][lane * 8] = acc2;
        *(f32x8*)&red[mt][h][3][lane * 8] = acc3;
    }
    __syncthreads();
    if (h != 0) return;

#pragma unroll
    for (int hh = 1; hh < KSPLIT; ++hh) {
        acc0 += *(const f32x8*)&red[mt][hh][0][lane * 8];
        acc1 += *(const f32x8*)&red[mt][hh][1][lane * 8];
        acc2 += *(const f32x8*)&red[mt][hh][2][lane * 8];
        acc3 += *(const f32x8*)&red[mt][hh][3][lane * 8];
    }

    // C-fragment: VGPR r -> row m0+r (lanes 0-15) / m0+r+8 (lanes 16-31),
    // expert = nt*16 + lane%16. Reductions via xor masks < 16 stay in-half.
    const float L2E = 1.44269504088896f;
#pragma unroll
    for (int r = 0; r < 8; ++r) {
        float v0 = acc0[r], v1 = acc1[r], v2 = acc2[r], v3 = acc3[r];

        float mx = fmaxf(fmaxf(v0, v1), fmaxf(v2, v3));
#pragma unroll
        for (int msk = 1; msk < 16; msk <<= 1)
            mx = fmaxf(mx, __shfl_xor(mx, msk, 32));

        float e0 = exp2f((v0 - mx) * L2E);
        float e1 = exp2f((v1 - mx) * L2E);
        float e2 = exp2f((v2 - mx) * L2E);
        float e3 = exp2f((v3 - mx) * L2E);
        float s = e0 + e1 + e2 + e3;
#pragma unroll
        for (int msk = 1; msk < 16; msk <<= 1)
            s += __shfl_xor(s, msk, 32);

        float inv = 1.0f / s;
        float p0 = e0 * inv, p1 = e1 * inv, p2 = e2 * inv, p3 = e3 * inv;

        const int token = m0 + r + (hi << 3);
        float* prow = probs + (size_t)token * NE + mrow;
        __builtin_nontemporal_store(p0, prow);
        __builtin_nontemporal_store(p1, prow + 16);
        __builtin_nontemporal_store(p2, prow + 32);
        __builtin_nontemporal_store(p3, prow + 48);

        // per-lane top-2 (ties -> lower index, matching lax.top_k)
        float b1v, b2v; int b1i, b2i;
        if (p0 >= p1) { b1v = p0; b1i = mrow;      b2v = p1; b2i = mrow + 16; }
        else          { b1v = p1; b1i = mrow + 16; b2v = p0; b2i = mrow; }
        top2_insert(p2, mrow + 32, b1v, b1i, b2v, b2i);
        top2_insert(p3, mrow + 48, b1v, b1i, b2v, b2i);

#pragma unroll
        for (int msk = 1; msk < 16; msk <<= 1) {
            float ov1 = __shfl_xor(b1v, msk, 32); int oi1 = __shfl_xor(b1i, msk, 32);
            float ov2 = __shfl_xor(b2v, msk, 32); int oi2 = __shfl_xor(b2i, msk, 32);
            top2_insert(ov1, oi1, b1v, b1i, b2v, b2i);
            top2_insert(ov2, oi2, b1v, b1i, b2v, b2i);
        }

        if (mrow == 0) {
            idx_out[token * 2 + 0] = (float)b1i;
            idx_out[token * 2 + 1] = (float)b2i;
            val_out[token * 2 + 0] = b1v;
            val_out[token * 2 + 1] = b2v;
        }
    }
}

extern "C" void kernel_launch(void* const* d_in, const int* in_sizes, int n_in,
                              void* d_out, int out_size, void* d_ws, size_t ws_size,
                              hipStream_t stream) {
    const float* x = (const float*)d_in[0];   // [B,S,D] fp32
    const float* W = (const float*)d_in[1];   // [E,D]   fp32
    (void)in_sizes; (void)n_in; (void)out_size; (void)ws_size;

    __bf16* Wb   = (__bf16*)d_ws;                       // E*D bf16 = 512 KB
    float* probs = (float*)d_out;                       // [B,S,E]
    float* idxo  = probs + (size_t)NTOK * NE;           // [B,S,2] as float
    float* valo  = idxo + (size_t)NTOK * 2;             // [B,S,2]

    wprep_kernel<<<(NE * DM) / 256, 256, 0, stream>>>(W, Wb);
    // 16384 tokens / (2 M-tiles * 16 tokens) = 512 blocks of 256 threads,
    // 4096 waves total (4-way K split) for latency hiding on the x stream.
    router_kernel<<<NTOK / (16 * MT_PER_BLOCK), 256, 0, stream>>>(x, Wb, probs, idxo, valo);
}
